// GGMLLayer_20229295964826
// MI455X (gfx1250) — matrix operand order repaired and checked by reading, not compiled, not run on hardware
//
#include <hip/hip_runtime.h>
#include <hip/hip_bf16.h>
#include <stdint.h>

typedef __attribute__((ext_vector_type(16))) __bf16 v16bf;
typedef __attribute__((ext_vector_type(8)))  __bf16 v8bf;
typedef __attribute__((ext_vector_type(8)))  float  v8f;

#define BATCH   256
#define IN_F    8192
#define OUT_F   8192
#define QK_K    256
#define BLK_B   144
#define NBLK_K  (IN_F / QK_K)

#define M_TILE  128
#define N_TILE  128
#define K_CHUNK 256
#define LDS_STR 264                       // halves per LDS row (256 + 8 pad)
#define NCHUNK  (IN_F / K_CHUNK)          // 32
#define A_HALVES   (M_TILE * LDS_STR)
#define BUF_HALVES ((M_TILE + N_TILE) * LDS_STR)   // one ping-pong buffer

#if defined(__gfx1250__)
#define HAVE_ASYNC 1
#else
#define HAVE_ASYNC 0
#endif

__device__ __forceinline__ uint32_t f2bf1(float f) {
    uint32_t u = __float_as_uint(f);
    u += 0x7FFFu + ((u >> 16) & 1u);      // round-to-nearest-even
    return u >> 16;
}

__device__ __forceinline__ uint32_t pack_bf16(float a, float b) {
#if defined(__AMDGCN__) && __has_builtin(__builtin_amdgcn_cvt_pk_bf16_f32)
    auto p = __builtin_amdgcn_cvt_pk_bf16_f32(a, b);
    return __builtin_bit_cast(uint32_t, p);
#elif defined(__gfx1250__)
    uint32_t r;
    asm("v_cvt_pk_bf16_f32 %0, %1, %2" : "=v"(r) : "v"(a), "v"(b));
    return r;
#else
    return f2bf1(a) | (f2bf1(b) << 16);
#endif
}

// Async DMA: global (16B) -> LDS, tracked by ASYNCcnt.
__device__ __forceinline__ void async_copy_b128(const void* g, void* l) {
#if HAVE_ASYNC
    uint32_t laddr = (uint32_t)(uintptr_t)l;
    uint64_t gaddr = (uint64_t)(uintptr_t)g;
    asm volatile("global_load_async_to_lds_b128 %0, %1, off"
                 :: "v"(laddr), "v"(gaddr)
                 : "memory");
#else
    *(uint4*)l = *(const uint4*)g;
#endif
}

__device__ __forceinline__ void wait_async_lds() {
#if HAVE_ASYNC
#if __has_builtin(__builtin_amdgcn_s_wait_asynccnt)
    __builtin_amdgcn_s_wait_asynccnt(0);
#else
    asm volatile("s_wait_asynccnt 0" ::: "memory");
#endif
#endif
}

// ---------------- kernel 0: x fp32 -> bf16 (workspace) ----------------
__global__ __launch_bounds__(256) void cvt_x_bf16(const float4* __restrict__ x4,
                                                  uint4* __restrict__ o4) {
    int i = blockIdx.x * blockDim.x + threadIdx.x;
    float4 a = x4[2 * i];
    float4 b = x4[2 * i + 1];
    uint4 r;
    r.x = pack_bf16(a.x, a.y);
    r.y = pack_bf16(a.z, a.w);
    r.z = pack_bf16(b.x, b.y);
    r.w = pack_bf16(b.z, b.w);
    o4[i] = r;
}

// Per-thread staged Q4_K data (half a block: 128 q-values)
struct BStage {
    uint4 hdr;                 // d|dmin (x), scales[0..3] (y), [4..7] (z), [8..11] (w)
    uint4 qv0, qv1, qv2, qv3;  // 64 q bytes
};

// Phase 1: issue all global traffic for chunk kc (no waits).
__device__ __forceinline__ void stage_load(
        const uint16_t* __restrict__ xb, const uint8_t* __restrict__ qw,
        uint16_t* __restrict__ Abuf, BStage& st,
        int m_base, int n_base, int kc, int tid)
{
    // A tile: 128 rows x 256 bf16; 2 threads/row, 16x b128 async copies
    {
        const int row = tid >> 1;
        const int seg = (tid & 1) * 128;
        const uint16_t* src = xb + (size_t)(m_base + row) * IN_F + kc + seg;
        uint16_t* dst = Abuf + row * LDS_STR + seg;
#pragma unroll
        for (int i = 0; i < 16; ++i)
            async_copy_b128(src + i * 8, dst + i * 8);
    }
    // B data: header (16B) + 64 q bytes into registers
    {
        const int nrow = tid >> 1;
        const int h    = tid & 1;
        const uint8_t* blk = qw + ((size_t)(n_base + nrow) * NBLK_K + (kc >> 8)) * BLK_B;
        st.hdr = *(const uint4*)blk;
        const uint4* qsp = (const uint4*)(blk + 16 + (h << 6));
        st.qv0 = qsp[0]; st.qv1 = qsp[1]; st.qv2 = qsp[2]; st.qv3 = qsp[3];
    }
}

// Phase 2: dequant (VALU) + LDS stores — runs under the WMMA shadow.
__device__ __forceinline__ void stage_math(
        uint16_t* __restrict__ Bbuf, const BStage& st, int tid)
{
    const int nrow = tid >> 1;
    const int h    = tid & 1;

    union { uint16_t u; _Float16 f; } cv;
    cv.u = (uint16_t)(st.hdr.x & 0xFFFFu);
    const float d = (float)cv.f;
    cv.u = (uint16_t)(st.hdr.x >> 16);
    const float dmin = (float)cv.f;

    uint8_t scales[12];
#pragma unroll
    for (int i = 0; i < 4; ++i) {
        scales[i]     = (st.hdr.y >> (8 * i)) & 0xFF;
        scales[i + 4] = (st.hdr.z >> (8 * i)) & 0xFF;
        scales[i + 8] = (st.hdr.w >> (8 * i)) & 0xFF;
    }

    const uint32_t q32[16] = { st.qv0.x, st.qv0.y, st.qv0.z, st.qv0.w,
                               st.qv1.x, st.qv1.y, st.qv1.z, st.qv1.w,
                               st.qv2.x, st.qv2.y, st.qv2.z, st.qv2.w,
                               st.qv3.x, st.qv3.y, st.qv3.z, st.qv3.w };

    uint16_t* brow = Bbuf + nrow * LDS_STR + 128 * h;
#pragma unroll
    for (int s4 = 0; s4 < 4; ++s4) {
        const int s = 4 * h + s4;          // global sub-block 0..7
        uint32_t sc, mn;
        if (s < 4) {
            sc = scales[s] & 63u;
            mn = scales[s + 4] & 63u;
        } else {
            sc = (scales[s + 4] & 15u) | ((scales[s - 4] >> 6) << 4);
            mn = (scales[s + 4] >> 4)  | ((scales[s] >> 6) << 4);
        }
        const float dl = d * (float)sc;
        const float ml = dmin * (float)mn;
        const int wbase = 8 * (s4 >> 1);
        const int nsh   = 4 * (s4 & 1);
#pragma unroll
        for (int dw = 0; dw < 8; ++dw) {
            uint32_t w = (q32[wbase + dw] >> nsh) & 0x0F0F0F0Fu;
            float f0 = (float)(w & 0xFFu);           // v_cvt_f32_ubyte0
            float f1 = (float)((w >> 8) & 0xFFu);
            float f2 = (float)((w >> 16) & 0xFFu);
            float f3 = (float)(w >> 24);
            uint2 p;
            p.x = pack_bf16(fmaf(f0, dl, -ml), fmaf(f1, dl, -ml));
            p.y = pack_bf16(fmaf(f2, dl, -ml), fmaf(f3, dl, -ml));
            *(uint2*)(brow + s4 * 32 + dw * 4) = p;  // ds_store_b64
        }
    }
}

// ---------------- kernel 1: double-buffered Q4_K bf16 WMMA GEMM ----------------
__global__ __launch_bounds__(256) void q4k_gemm_wmma(
        const uint16_t* __restrict__ xb,
        const uint8_t*  __restrict__ qw,
        const float*    __restrict__ bias,
        float*          __restrict__ out)
{
    extern __shared__ char smem[];
    uint16_t* smem_h = (uint16_t*)smem;

    const int tid  = threadIdx.x;
    const int wave = tid >> 5;
    const int lane = tid & 31;
    const int half = lane >> 4;
    const int l16  = lane & 15;

    const int n_base = blockIdx.x * N_TILE;
    const int m_base = blockIdx.y * M_TILE;

    const int wm = (wave & 1) * 64;
    const int wn = (wave >> 1) * 32;

    v8f acc[4][2];
#pragma unroll
    for (int i = 0; i < 4; ++i)
#pragma unroll
        for (int j = 0; j < 2; ++j) acc[i][j] = (v8f){0.f,0.f,0.f,0.f,0.f,0.f,0.f,0.f};

    BStage st;

    // prologue: fully stage chunk 0 into buffer 0
    stage_load(xb, qw, smem_h, st, m_base, n_base, 0, tid);
    stage_math(smem_h + A_HALVES, st, tid);

    for (int c = 0; c < NCHUNK; ++c) {
        uint16_t* Acur = smem_h + (c & 1) * BUF_HALVES;
        uint16_t* Bcur = Acur + A_HALVES;
        uint16_t* Anxt = smem_h + ((c + 1) & 1) * BUF_HALVES;
        const bool have_next = (c + 1 < NCHUNK);

        wait_async_lds();      // our async copies for buf[c&1] done
        __syncthreads();       // everyone's stores for buf[c&1] visible

        // issue next chunk's global traffic before the matrix work
        if (have_next)
            stage_load(xb, qw, Anxt, st, m_base, n_base, (c + 1) * K_CHUNK, tid);

        // ---- 8 WMMA k-steps of 32 over this K chunk ----
#pragma unroll
        for (int kk = 0; kk < K_CHUNK; kk += 32) {
            v16bf afrag[4];
#pragma unroll
            for (int i = 0; i < 4; ++i) {
                const uint16_t* ap = Acur + (wm + i * 16 + l16) * LDS_STR + kk;
                v8bf lo = *(const v8bf*)(ap + 16 * half);
                v8bf hi = *(const v8bf*)(ap + 16 * half + 8);
                afrag[i] = __builtin_shufflevector(lo, hi,
                    0, 1, 2, 3, 4, 5, 6, 7, 8, 9, 10, 11, 12, 13, 14, 15);
            }
            v16bf bfrag[2];
#pragma unroll
            for (int j = 0; j < 2; ++j) {
                const uint16_t* bp = Bcur + (wn + j * 16 + l16) * LDS_STR + kk + 16 * half;
                bfrag[j] = *(const v16bf*)bp;
            }
#pragma unroll
            for (int i = 0; i < 4; ++i)
#pragma unroll
                for (int j = 0; j < 2; ++j)
                    acc[i][j] = __builtin_amdgcn_wmma_f32_16x16x32_bf16(
                        false, afrag[i], false, bfrag[j],
                        (short)0, acc[i][j], false, false);
        }

        // dequant math for next chunk — co-executes under XDL WMMA shadow
        if (have_next)
            stage_math(Anxt + A_HALVES, st, tid);
    }

    // ---- epilogue: C layout — VGPR r: M = r + 8*half, N = lane&15 ----
#pragma unroll
    for (int j = 0; j < 2; ++j) {
        const int n = n_base + wn + j * 16 + l16;
        const float bv = bias[n];
#pragma unroll
        for (int i = 0; i < 4; ++i) {
            const int m0 = m_base + wm + i * 16 + 8 * half;
#pragma unroll
            for (int r = 0; r < 8; ++r) {
                out[(size_t)(m0 + r) * OUT_F + n] = acc[i][j][r] + bv;
            }
        }
    }
}

extern "C" void kernel_launch(void* const* d_in, const int* in_sizes, int n_in,
                              void* d_out, int out_size, void* d_ws, size_t ws_size,
                              hipStream_t stream) {
    const float*   x    = (const float*)d_in[0];
    const uint8_t* qw   = (const uint8_t*)d_in[1];
    const float*   bias = (const float*)d_in[2];
    float*         out  = (float*)d_out;
    uint16_t*      xb   = (uint16_t*)d_ws;     // BATCH*IN_F bf16 = 4 MB

    const int n_elems = BATCH * IN_F;
    cvt_x_bf16<<<n_elems / (256 * 8), 256, 0, stream>>>((const float4*)x, (uint4*)xb);

    dim3 grid(OUT_F / N_TILE, BATCH / M_TILE); // (64, 2)
    size_t smem = (size_t)2 * BUF_HALVES * 2;  // 2 buffers, 264 KB
    q4k_gemm_wmma<<<grid, 256, smem, stream>>>(xb, qw, bias, out);
}
